// MambaBlock_13838384628222
// MI455X (gfx1250) — matrix operand order repaired and checked by reading, not compiled, not run on hardware
//
#include <hip/hip_runtime.h>

// ---------------------------------------------------------------------------
// Mamba block for MI455X (gfx1250, wave32, WMMA).
// Stage 0: one-shot f32->bf16 conversion of x and transposed bf16 copies of
//          W_in/W_dt/W_out, so every GEMM operand tile is a contiguous bf16
//          row block (async-copy friendly).
// GEMMs:   128x128x32 tiles, 8 waves, v_wmma_f32_16x16x32_bf16, double-
//          buffered LDS staged with GLOBAL_LOAD_ASYNC_TO_LDS_B128 when the
//          toolchain exposes it (ASYNCcnt), else synchronous b128 copies.
// Scan:    per-lane 16-state recurrence (VALU + v_exp_f32), fused out-gate.
// ---------------------------------------------------------------------------

typedef __bf16 bf16;
typedef __attribute__((ext_vector_type(16))) __bf16 v16bf;
typedef __attribute__((ext_vector_type(8)))  __bf16 v8bf;
typedef __attribute__((ext_vector_type(8)))  float  v8f;
typedef __attribute__((ext_vector_type(4)))  float  v4f;
typedef __attribute__((ext_vector_type(4)))  int    v4i;

#define D_MODEL 1024
#define D_STATE 16
#define D_INNER 2048
#define BATCH   8
#define SEQ     2048
#define MROWS   (BATCH * SEQ)          // 16384 rows in all GEMMs

// GEMM tiling: 256 threads = 8 waves; block tile 128x128, K-step 32.
// Wave grid 2 (M) x 4 (N); each wave owns a 64x32 region = 4x2 WMMA tiles.
#define BM 128
#define BN 128
#define BK 32
#define LDT 40   // LDS tile row pitch (32 + 8 pad bf16; 80 B = 5*16 B aligned)

enum { EP_SPLIT_SILU = 0, EP_SOFTPLUS = 1, EP_BIAS_F32 = 2 };

#if defined(__has_builtin)
#if __has_builtin(__builtin_amdgcn_global_load_async_to_lds_b128)
#define USE_ASYNC_LDS 1
#endif
#endif
#ifndef USE_ASYNC_LDS
#define USE_ASYNC_LDS 0
#endif

__device__ __forceinline__ float siluf(float x) {
  // x * sigmoid(x) with v_rcp_f32 (avoid the IEEE divide expansion)
  return x * __builtin_amdgcn_rcpf(1.0f + __expf(-x));
}
__device__ __forceinline__ float softplusf(float x) {
  return (x > 20.0f) ? x : __logf(1.0f + __expf(x));
}

// 16-byte global -> LDS copy, async when available.
// Builtin signature (from compiler diagnostic): first param is int4 in the
// global (__device__) address space, second is the LDS destination.
__device__ __forceinline__ void cp16_g2l(const bf16* __restrict__ src,
                                         bf16* __restrict__ dst) {
#if USE_ASYNC_LDS
  __builtin_amdgcn_global_load_async_to_lds_b128(
      (__attribute__((address_space(1))) v4i*)src,
      (__attribute__((address_space(3))) v4i*)dst, 0, 0);
#else
  *(v8bf*)dst = *(const v8bf*)src;
#endif
}

__device__ __forceinline__ void wait_g2l() {
#if USE_ASYNC_LDS
#if __has_builtin(__builtin_amdgcn_s_wait_asynccnt)
  __builtin_amdgcn_s_wait_asynccnt(0);
#else
  asm volatile("s_wait_asynccnt 0x0" ::: "memory");
#endif
#endif
}

// Stage one 128x32 bf16 tile (row pitch K in global, LDT in LDS).
// 128 rows * 4 chunks = 512 16B chunks; 256 threads -> 2 chunks each.
__device__ __forceinline__ void stage_tile(const bf16* __restrict__ g, int K,
                                           bf16* __restrict__ l, int tid) {
#pragma unroll
  for (int i = 0; i < 2; ++i) {
    const int id = tid + i * 256;
    const int r  = id >> 2;
    const int c  = (id & 3) << 3;
    cp16_g2l(g + (size_t)r * K + c, l + r * LDT + c);
  }
}

// ---------------------------------------------------------------------------
// Tiled bf16 WMMA GEMM: out = epilogue(Abf[M,K] @ WT[N,K]^T + bias).
// Both operands bf16; WT is pre-transposed so its tiles are contiguous.
// ---------------------------------------------------------------------------
template <int EP>
__global__ __launch_bounds__(256) void gemm_bf16_wmma(
    const bf16* __restrict__ Abf, const bf16* __restrict__ WT,
    const float* __restrict__ bias, int K, int N,
    void* __restrict__ out0, void* __restrict__ out1)
{
  __shared__ __attribute__((aligned(16))) bf16 ldsA[2][BM * LDT];
  __shared__ __attribute__((aligned(16))) bf16 ldsB[2][BN * LDT];

  const int tid  = threadIdx.x;
  const int lane = tid & 31;
  const int wid  = tid >> 5;
  const int wm   = wid >> 2;        // 0..1  (64-row slab)
  const int wn   = wid & 3;         // 0..3  (32-col slab)
  const int half = lane >> 4;       // lane group (ISA A/B layout)
  const int l16  = lane & 15;

  const int n0 = blockIdx.x * BN;
  const int m0 = blockIdx.y * BM;

  const bf16* gA = Abf + (size_t)m0 * K;   // A tile rows
  const bf16* gB = WT  + (size_t)n0 * K;   // B tile rows (pre-transposed)

  const v8f vzero = {0.f, 0.f, 0.f, 0.f, 0.f, 0.f, 0.f, 0.f};
  v8f acc[4][2];
#pragma unroll
  for (int i = 0; i < 4; ++i)
#pragma unroll
    for (int j = 0; j < 2; ++j) acc[i][j] = vzero;

  // prologue: stage tile 0
  stage_tile(gA, K, ldsA[0], tid);
  stage_tile(gB, K, ldsB[0], tid);
  wait_g2l();
  __syncthreads();

  const int KT = K / BK;
  for (int kt = 0; kt < KT; ++kt) {
    const int cur = kt & 1;
    if (kt + 1 < KT) {  // stage next tile into the other buffer
      stage_tile(gA + (size_t)(kt + 1) * BK, K, ldsA[cur ^ 1], tid);
      stage_tile(gB + (size_t)(kt + 1) * BK, K, ldsB[cur ^ 1], tid);
    }

    // ---- B fragments: lane holds col N=l16, K = 16*half .. 16*half+15 ----
    v16bf bfrag[2];
#pragma unroll
    for (int tn = 0; tn < 2; ++tn) {
      const bf16* pb = &ldsB[cur][(wn * 32 + tn * 16 + l16) * LDT + 16 * half];
      const v8bf blo = *(const v8bf*)(pb);
      const v8bf bhi = *(const v8bf*)(pb + 8);
      bfrag[tn] = __builtin_shufflevector(
          blo, bhi, 0, 1, 2, 3, 4, 5, 6, 7, 8, 9, 10, 11, 12, 13, 14, 15);
    }
    // ---- A fragments + WMMA: lane row M=l16; K = 16*half .. 16*half+15 ----
#pragma unroll
    for (int tm = 0; tm < 4; ++tm) {
      const bf16* pa = &ldsA[cur][(wm * 64 + tm * 16 + l16) * LDT + 16 * half];
      const v8bf alo = *(const v8bf*)(pa);
      const v8bf ahi = *(const v8bf*)(pa + 8);
      const v16bf afrag = __builtin_shufflevector(
          alo, ahi, 0, 1, 2, 3, 4, 5, 6, 7, 8, 9, 10, 11, 12, 13, 14, 15);
#pragma unroll
      for (int tn = 0; tn < 2; ++tn) {
        acc[tm][tn] = __builtin_amdgcn_wmma_f32_16x16x32_bf16(
            /*neg_a=*/false, afrag, /*neg_b=*/false, bfrag[tn],
            /*c_mod=*/(short)0, acc[tm][tn],
            /*reuse_a=*/false, /*reuse_b=*/false);
      }
    }
    wait_g2l();        // next-buffer async writes complete
    __syncthreads();   // ..and everyone is done reading the current buffer
  }

  // ---- epilogue: C/D layout -> lane col = l16, VGPR r -> row r + 8*half ----
#pragma unroll
  for (int tm = 0; tm < 4; ++tm) {
#pragma unroll
    for (int tn = 0; tn < 2; ++tn) {
      const int col = n0 + wn * 32 + tn * 16 + l16;
      const float bc = bias[col];
#pragma unroll
      for (int r = 0; r < 8; ++r) {
        const int row = m0 + wm * 64 + tm * 16 + half * 8 + r;
        const float v = acc[tm][tn][r] + bc;
        if (EP == EP_SPLIT_SILU) {
          // xz split: cols [0,2048) -> u = silu(x_proj); [2048,4096) -> silu(z)
          const float s = siluf(v);
          if (col < D_INNER)
            ((bf16*)out0)[(size_t)row * D_INNER + col] = (bf16)s;
          else
            ((bf16*)out1)[(size_t)row * D_INNER + (col - D_INNER)] = (bf16)s;
        } else if (EP == EP_SOFTPLUS) {
          ((bf16*)out0)[(size_t)row * N + col] = (bf16)softplusf(v);
        } else {  // EP_BIAS_F32: final projection, f32 output
          ((float*)out0)[(size_t)row * N + col] = v;
        }
      }
    }
  }
}

// ---------------------------------------------------------------------------
// Elementwise f32 -> bf16 (for x).
// ---------------------------------------------------------------------------
__global__ __launch_bounds__(256) void cvt_bf16_kernel(
    const float* __restrict__ in, bf16* __restrict__ out)
{
  const int i = (blockIdx.x * 256 + threadIdx.x) * 4;
  const v4f v = *(const v4f*)(in + i);
  bf16* o = out + i;
  o[0] = (bf16)v[0]; o[1] = (bf16)v[1]; o[2] = (bf16)v[2]; o[3] = (bf16)v[3];
}

// ---------------------------------------------------------------------------
// Transpose + convert: in f32 [K][N] -> out bf16 [N][K].  32x32 LDS tiles.
// ---------------------------------------------------------------------------
__global__ __launch_bounds__(256) void transpose_cvt_kernel(
    const float* __restrict__ in, bf16* __restrict__ out, int K, int N)
{
  __shared__ float t[32][33];
  const int tx = threadIdx.x & 31;
  const int ty = threadIdx.x >> 5;   // 0..7
  const int n0 = blockIdx.x * 32;
  const int k0 = blockIdx.y * 32;
#pragma unroll
  for (int i = 0; i < 4; ++i) {
    const int k = ty + i * 8;
    t[k][tx] = in[(size_t)(k0 + k) * N + (n0 + tx)];
  }
  __syncthreads();
#pragma unroll
  for (int i = 0; i < 4; ++i) {
    const int n = ty + i * 8;
    out[(size_t)(n0 + n) * K + (k0 + tx)] = (bf16)t[tx][n];
  }
}

// ---------------------------------------------------------------------------
// x_dbl = u @ W_x + b_x   (N = 32: one wave per row, lane j = output column).
// ---------------------------------------------------------------------------
__global__ __launch_bounds__(256) void xproj_kernel(
    const bf16* __restrict__ u, const float* __restrict__ Wx,
    const float* __restrict__ bx, float* __restrict__ xdbl)
{
  const int row = blockIdx.x * 8 + (threadIdx.x >> 5);
  const int j   = threadIdx.x & 31;
  const bf16* ur = u + (size_t)row * D_INNER;
  float acc = 0.0f;
  for (int k = 0; k < D_INNER; k += 4) {
    const float u0 = (float)ur[k + 0];
    const float u1 = (float)ur[k + 1];
    const float u2 = (float)ur[k + 2];
    const float u3 = (float)ur[k + 3];
    acc += u0 * Wx[(size_t)(k + 0) * 32 + j];
    acc += u1 * Wx[(size_t)(k + 1) * 32 + j];
    acc += u2 * Wx[(size_t)(k + 2) * 32 + j];
    acc += u3 * Wx[(size_t)(k + 3) * 32 + j];
  }
  xdbl[(size_t)row * 32 + j] = acc + bx[j];
}

// ---------------------------------------------------------------------------
// Selective scan: one lane per (batch, channel). 16-state recurrence in
// registers, v_exp_f32 for dA. Fuses the output gate y * silu(z) and writes
// g in place over u (each lane reads u[idx] before writing g[idx]).
// ---------------------------------------------------------------------------
__global__ __launch_bounds__(256) void scan_kernel(
    bf16* __restrict__ u_g, const bf16* __restrict__ dt,
    const float* __restrict__ xdbl, const bf16* __restrict__ zs,
    const float* __restrict__ A_log, const float* __restrict__ Dp)
{
  const int gid = blockIdx.x * blockDim.x + threadIdx.x;   // 0..16383
  const int b = gid / D_INNER;
  const int d = gid - b * D_INNER;

  float A[D_STATE], h[D_STATE];
#pragma unroll
  for (int s = 0; s < D_STATE; ++s) {
    A[s] = -__expf(A_log[(size_t)d * D_STATE + s]);
    h[s] = 0.0f;
  }
  const float Dv = Dp[d];
  const size_t rowBase = (size_t)b * SEQ;

  for (int t = 0; t < SEQ; ++t) {
    const size_t row = rowBase + t;
    const size_t idx = row * D_INNER + d;
    const float dtv = (float)dt[idx];
    const float uv  = (float)u_g[idx];
    const float* pbc = xdbl + row * 32;   // [B(16) | C(16)], wave-broadcast
    const float du = dtv * uv;
    float y = 0.0f;
#pragma unroll
    for (int s = 0; s < D_STATE; ++s) {
      const float dA = __expf(dtv * A[s]);
      h[s] = h[s] * dA + du * pbc[s];
      y += h[s] * pbc[D_STATE + s];
    }
    const float g = (y + Dv * uv) * (float)zs[idx];
    u_g[idx] = (bf16)g;
  }
}

// ---------------------------------------------------------------------------
// Launch. Inputs (setup_inputs order):
// 0:x 1:W_in 2:b_in 3:W_x 4:b_x 5:W_dt 6:b_dt 7:W_out 8:b_out 9:A_log 10:D
// Workspace layout (bytes):
//   [  0,  64M)  u (bf16)        -> reused in-place as g after the scan
//   [ 64M,128M)  silu(z) (bf16)
//   [128M,192M)  dt (bf16, post-softplus)
//   [192M,194M)  x_dbl (f32 [M,32])
//   [194M,226M)  x (bf16 [M,1024])
//   [226M,234M)  W_in^T  (bf16 [4096,1024])
//   [234M,242M)  W_dt^T  (bf16 [2048,2048])
//   [242M,246M)  W_out^T (bf16 [1024,2048])
// ---------------------------------------------------------------------------
extern "C" void kernel_launch(void* const* d_in, const int* in_sizes, int n_in,
                              void* d_out, int out_size, void* d_ws, size_t ws_size,
                              hipStream_t stream) {
  const float* x     = (const float*)d_in[0];
  const float* W_in  = (const float*)d_in[1];
  const float* b_in  = (const float*)d_in[2];
  const float* W_x   = (const float*)d_in[3];
  const float* b_x   = (const float*)d_in[4];
  const float* W_dt  = (const float*)d_in[5];
  const float* b_dt  = (const float*)d_in[6];
  const float* W_out = (const float*)d_in[7];
  const float* b_out = (const float*)d_in[8];
  const float* A_log = (const float*)d_in[9];
  const float* D_par = (const float*)d_in[10];

  char* ws = (char*)d_ws;
  size_t off = 0;
  bf16*  u    = (bf16*)(ws + off); off += (size_t)MROWS * D_INNER * 2;    // 64M
  bf16*  zs   = (bf16*)(ws + off); off += (size_t)MROWS * D_INNER * 2;    // 64M
  bf16*  dt   = (bf16*)(ws + off); off += (size_t)MROWS * D_INNER * 2;    // 64M
  float* xd   = (float*)(ws + off); off += (size_t)MROWS * 32 * 4;        //  2M
  bf16*  xbf  = (bf16*)(ws + off); off += (size_t)MROWS * D_MODEL * 2;    // 32M
  bf16*  WinT = (bf16*)(ws + off); off += (size_t)(2 * D_INNER) * D_MODEL * 2;
  bf16*  WdtT = (bf16*)(ws + off); off += (size_t)D_INNER * D_INNER * 2;
  bf16*  WoutT= (bf16*)(ws + off); off += (size_t)D_MODEL * D_INNER * 2;
  (void)ws_size; (void)in_sizes; (void)n_in; (void)out_size;

  const dim3 blk(256);

  // 0) one-shot operand conversion / weight transposition (bf16)
  cvt_bf16_kernel<<<dim3((MROWS * D_MODEL) / 1024), blk, 0, stream>>>(x, xbf);
  transpose_cvt_kernel<<<dim3((2 * D_INNER) / 32, D_MODEL / 32), blk, 0, stream>>>(
      W_in, WinT, D_MODEL, 2 * D_INNER);
  transpose_cvt_kernel<<<dim3(D_INNER / 32, D_INNER / 32), blk, 0, stream>>>(
      W_dt, WdtT, D_INNER, D_INNER);
  transpose_cvt_kernel<<<dim3(D_MODEL / 32, D_INNER / 32), blk, 0, stream>>>(
      W_out, WoutT, D_INNER, D_MODEL);

  // 1) xz = x @ W_in + b_in, split + SiLU -> u, silu(z)
  gemm_bf16_wmma<EP_SPLIT_SILU>
      <<<dim3((2 * D_INNER) / BN, MROWS / BM), blk, 0, stream>>>(
          xbf, WinT, b_in, D_MODEL, 2 * D_INNER, u, zs);

  // 2) x_dbl = u @ W_x + b_x   ([M,32] = [B|C])
  xproj_kernel<<<dim3(MROWS / 8), blk, 0, stream>>>(u, W_x, b_x, xd);

  // 3) dt = softplus(u @ W_dt + b_dt)
  gemm_bf16_wmma<EP_SOFTPLUS>
      <<<dim3(D_INNER / BN, MROWS / BM), blk, 0, stream>>>(
          u, WdtT, b_dt, D_INNER, D_INNER, dt, nullptr);

  // 4) selective scan + output gate -> g (in place over u)
  scan_kernel<<<dim3((BATCH * D_INNER) / 256), blk, 0, stream>>>(
      u, dt, xd, zs, A_log, D_par);

  // 5) out = g @ W_out + b_out  (f32)
  gemm_bf16_wmma<EP_BIAS_F32>
      <<<dim3(D_MODEL / BN, MROWS / BM), blk, 0, stream>>>(
          u, WoutT, b_out, D_INNER, D_MODEL, d_out, nullptr);
}
